// Encoder_51900384804901
// MI455X (gfx1250) — compile-verified, matrix operand order checked by reading, not run on hardware
//
#include <hip/hip_runtime.h>
#include <hip/hip_bf16.h>

// ---------------------------------------------------------------------------
// Transformer encoder (L=6, E=1024, NH=16, DH=64, DFF=4096, B=8, S=2048, NC=1024)
// for MI455X / gfx1250: wave32, WMMA f32_16x16x32_f16, LDS-tiled GEMMs,
// flash-attention with online softmax in WMMA register layout.
//
// GEMM: block tile 256x128, 8 waves in 4x2 grid, wave tile 64x64 (4x4 WMMA),
// K staged 64 deep per barrier -> 1 ds_load_b128 per WMMA (vs 1.5 at 64x32).
// ---------------------------------------------------------------------------

typedef __attribute__((ext_vector_type(16))) _Float16 v16h;
typedef __attribute__((ext_vector_type(8)))  float    v8f;

#define E_    1024
#define NH_   16
#define DH_   64
#define DFF_  4096
#define L_    6
#define B_    8
#define S_    2048
#define NC_   1024
#define MTOK  (B_ * S_)   // 16384 query tokens
#define MCTX  (B_ * NC_)  //  8192 context tokens

union FragU {
  v16h  v;
  uint4 u[2];
};

__device__ __forceinline__ v8f wmma_f16(const FragU& a, const FragU& b, v8f c) {
  // D = A(16x32 f16) x B(32x16 f16) + C(16x16 f32)
  return __builtin_amdgcn_wmma_f32_16x16x32_f16(
      /*neg_a=*/false, a.v, /*neg_b=*/false, b.v,
      /*c_mod=*/(short)0, c, /*reuse_a=*/false, /*reuse_b=*/false);
}

// ---------------------------------------------------------------------------
// Tiled GEMM:  out[M,N] = A[M,K](f16) * W[N,K](f16)^T + bias[N]
// rowDiv/rowMul remap logical row r -> global A row (r/rowDiv)*rowMul + r%rowDiv
// (used to read only the first NC rows per batch for K/V projections).
// ---------------------------------------------------------------------------
enum { ST_F32 = 0, ST_F16 = 1, ST_F16_RELU = 2, ST_VT = 3 };

template <int MODE>
__global__ __launch_bounds__(256) void gemm_kernel(
    const _Float16* __restrict__ A, const _Float16* __restrict__ W,
    const float* __restrict__ bias, float* __restrict__ outF,
    _Float16* __restrict__ outH, int M, int N, int K,
    int rowDiv, int rowMul, float scale)
{
  __shared__ _Float16 As[256][72];  // 64 used halves + 8 pad (144B stride)
  __shared__ _Float16 Ws[128][72];

  const int tid  = threadIdx.x;
  const int lane = tid & 31;
  const int wave = tid >> 5;
  const int wm   = wave >> 1;      // 4 waves along M (64 rows each) -> 256
  const int wn   = wave & 1;       // 2 waves along N (64 cols each) -> 128
  const int bm   = blockIdx.y * 256;
  const int bn   = blockIdx.x * 128;
  const int g    = lane >> 4;      // lane group (0/1)
  const int ln16 = lane & 15;
  const int co   = g << 3;         // fragment K-chunk offset (halves)

  v8f acc[4][4];
#pragma unroll
  for (int i = 0; i < 4; ++i)
#pragma unroll
    for (int j = 0; j < 4; ++j)
#pragma unroll
      for (int v = 0; v < 8; ++v) acc[i][j][v] = 0.f;

  const int nk = K >> 6;           // 64-deep K stages
  for (int kb = 0; kb < nk; ++kb) {
    __syncthreads();
    // Stage A tile (256 rows x 64 halves) and W tile (128 x 64) via b128.
#pragma unroll
    for (int it = 0; it < 8; ++it) {
      int idx = tid + it * 256;
      int row = idx >> 3;
      int c8  = (idx & 7) << 3;    // column offset in halves
      int r   = bm + row;
      int gr  = (r / rowDiv) * rowMul + (r % rowDiv);
      *(uint4*)&As[row][c8] =
          *(const uint4*)(A + (size_t)gr * K + (kb << 6) + c8);
      if (kb + 1 < nk)
        __builtin_prefetch(A + (size_t)gr * K + ((kb + 1) << 6) + c8, 0, 0);
    }
#pragma unroll
    for (int it = 0; it < 4; ++it) {
      int idx = tid + it * 256;
      int row = idx >> 3;
      int c8  = (idx & 7) << 3;
      *(uint4*)&Ws[row][c8] =
          *(const uint4*)(W + (size_t)(bn + row) * K + (kb << 6) + c8);
      if (kb + 1 < nk)
        __builtin_prefetch(W + (size_t)(bn + row) * K + ((kb + 1) << 6) + c8, 0, 0);
    }
    __syncthreads();

#pragma unroll
    for (int kc = 0; kc < 2; ++kc) {
      FragU a[4], w[4];
#pragma unroll
      for (int i = 0; i < 4; ++i) {
        int r = wm * 64 + i * 16 + ln16;     // A: lane = M row
        a[i].u[0] = *(const uint4*)&As[r][kc * 32 + co];
        a[i].u[1] = *(const uint4*)&As[r][kc * 32 + co + 16];
      }
#pragma unroll
      for (int j = 0; j < 4; ++j) {
        int r = wn * 64 + j * 16 + ln16;     // B: lane = N col (W row-major = B col-major)
        w[j].u[0] = *(const uint4*)&Ws[r][kc * 32 + co];
        w[j].u[1] = *(const uint4*)&Ws[r][kc * 32 + co + 16];
      }
#pragma unroll
      for (int i = 0; i < 4; ++i)
#pragma unroll
        for (int j = 0; j < 4; ++j)
          acc[i][j] = wmma_f16(a[i], w[j], acc[i][j]);
    }
  }

  // Epilogue: D layout -> element (v, lane): row = v + 8*(lane/16), col = lane%16
#pragma unroll
  for (int i = 0; i < 4; ++i) {
#pragma unroll
    for (int j = 0; j < 4; ++j) {
      int   c  = bn + wn * 64 + j * 16 + ln16;
      float bv = bias[c];
#pragma unroll
      for (int v = 0; v < 8; ++v) {
        int   r   = bm + wm * 64 + i * 16 + g * 8 + v;
        float val = acc[i][j][v] + bv;
        if constexpr (MODE == ST_F32) {
          outF[(size_t)r * N + c] = val;
        } else if constexpr (MODE == ST_F16) {
          outH[(size_t)r * N + c] = (_Float16)(val * scale);
        } else if constexpr (MODE == ST_F16_RELU) {
          outH[(size_t)r * N + c] = (_Float16)(val > 0.f ? val : 0.f);
        } else {  // ST_VT: V projection stored transposed: [B, NH, DH, NC]
          int bb = r / NC_, t = r % NC_;
          int hh = c >> 6, d = c & 63;
          outH[(((size_t)bb * NH_ + hh) * DH_ + d) * NC_ + t] = (_Float16)val;
        }
      }
    }
  }
}

// ---------------------------------------------------------------------------
// Flash attention: one block per (b, h, 64-row q tile), 4 waves.
// Q pre-scaled by 1/sqrt(DH). K in [B,NC,E] natural (= B col-major),
// V stored transposed [B,NH,DH,NC] (= B col-major for PV).
// Online softmax done in WMMA D layout; row stats reduced with shfl_xor
// butterflies over 16-lane halves (wave32).
// ---------------------------------------------------------------------------
__global__ __launch_bounds__(128) void flash_kernel(
    const _Float16* __restrict__ Q, const _Float16* __restrict__ Km,
    const _Float16* __restrict__ Vt, _Float16* __restrict__ O)
{
  __shared__ _Float16 Qs[64][72];
  __shared__ _Float16 Ks[64][72];
  __shared__ _Float16 Vs[64][72];
  __shared__ _Float16 Ps[64][72];

  const int tid  = threadIdx.x;
  const int lane = tid & 31;
  const int wave = tid >> 5;       // wave owns q rows [wave*16, wave*16+16)
  const int qt   = blockIdx.x;
  const int h    = blockIdx.y;
  const int b    = blockIdx.z;
  const int g    = lane >> 4;
  const int ln16 = lane & 15;
  const int co   = g << 3;

  // Load Q tile (64x64 halves)
#pragma unroll
  for (int it = 0; it < 4; ++it) {
    int idx = tid + it * 128;
    int row = idx >> 3, c8 = (idx & 7) << 3;
    *(uint4*)&Qs[row][c8] =
        *(const uint4*)(Q + (size_t)(b * S_ + qt * 64 + row) * E_ + h * DH_ + c8);
  }

  float m_old[8], l_old[8];
  v8f   o_acc[4];
#pragma unroll
  for (int v = 0; v < 8; ++v) { m_old[v] = -1e30f; l_old[v] = 0.f; }
#pragma unroll
  for (int tn = 0; tn < 4; ++tn)
#pragma unroll
    for (int v = 0; v < 8; ++v) o_acc[tn][v] = 0.f;

  for (int kb = 0; kb < NC_ / 64; ++kb) {
    __syncthreads();
#pragma unroll
    for (int it = 0; it < 4; ++it) {
      int idx = tid + it * 128;
      int row = idx >> 3, c8 = (idx & 7) << 3;
      *(uint4*)&Ks[row][c8] =
          *(const uint4*)(Km + (size_t)(b * NC_ + kb * 64 + row) * E_ + h * DH_ + c8);
      *(uint4*)&Vs[row][c8] =
          *(const uint4*)(Vt + (size_t)((b * NH_ + h) * DH_ + row) * NC_ + kb * 64 + c8);
    }
    __syncthreads();

    // S tile (16 q-rows x 64 keys per wave) = Q * K^T
    v8f sacc[4];
#pragma unroll
    for (int tn = 0; tn < 4; ++tn)
#pragma unroll
      for (int v = 0; v < 8; ++v) sacc[tn][v] = 0.f;

#pragma unroll
    for (int kc = 0; kc < 2; ++kc) {
      FragU aq;
      int   ar = wave * 16 + ln16;
      aq.u[0] = *(const uint4*)&Qs[ar][kc * 32 + co];
      aq.u[1] = *(const uint4*)&Qs[ar][kc * 32 + co + 16];
#pragma unroll
      for (int tn = 0; tn < 4; ++tn) {
        FragU bk;
        int   br = tn * 16 + ln16;  // key index = N column
        bk.u[0] = *(const uint4*)&Ks[br][kc * 32 + co];
        bk.u[1] = *(const uint4*)&Ks[br][kc * 32 + co + 16];
        sacc[tn] = wmma_f16(aq, bk, sacc[tn]);
      }
    }

    // Online softmax update (per D-layout row v + 8*g)
    float p[4][8];
#pragma unroll
    for (int v = 0; v < 8; ++v) {
      float mx = sacc[0][v];
#pragma unroll
      for (int tn = 1; tn < 4; ++tn) mx = fmaxf(mx, sacc[tn][v]);
#pragma unroll
      for (int msk = 1; msk <= 8; msk <<= 1)
        mx = fmaxf(mx, __shfl_xor(mx, msk, 32));
      float mn    = fmaxf(m_old[v], mx);
      float alpha = __expf(m_old[v] - mn);
      float rs    = 0.f;
#pragma unroll
      for (int tn = 0; tn < 4; ++tn) {
        float e = __expf(sacc[tn][v] - mn);
        p[tn][v] = e;
        rs += e;
      }
#pragma unroll
      for (int msk = 1; msk <= 8; msk <<= 1) rs += __shfl_xor(rs, msk, 32);
      l_old[v] = l_old[v] * alpha + rs;
      m_old[v] = mn;
#pragma unroll
      for (int tn = 0; tn < 4; ++tn) o_acc[tn][v] *= alpha;
    }

    // Scatter P (D layout) -> LDS row-major f16; wave reads only its own rows.
#pragma unroll
    for (int tn = 0; tn < 4; ++tn)
#pragma unroll
      for (int v = 0; v < 8; ++v)
        Ps[wave * 16 + g * 8 + v][tn * 16 + ln16] = (_Float16)p[tn][v];

    // O += P * V  (A = P rows, B = V^T rows = d)
#pragma unroll
    for (int kc = 0; kc < 2; ++kc) {
      FragU ap;
      int   ar = wave * 16 + ln16;
      ap.u[0] = *(const uint4*)&Ps[ar][kc * 32 + co];
      ap.u[1] = *(const uint4*)&Ps[ar][kc * 32 + co + 16];
#pragma unroll
      for (int tn = 0; tn < 4; ++tn) {
        FragU bv;
        int   br = tn * 16 + ln16;  // d index = N column
        bv.u[0] = *(const uint4*)&Vs[br][kc * 32 + co];
        bv.u[1] = *(const uint4*)&Vs[br][kc * 32 + co + 16];
        o_acc[tn] = wmma_f16(ap, bv, o_acc[tn]);
      }
    }
  }

  // Normalize and store f16 attention output [B,S,E]
#pragma unroll
  for (int tn = 0; tn < 4; ++tn)
#pragma unroll
    for (int v = 0; v < 8; ++v) {
      int r = qt * 64 + wave * 16 + g * 8 + v;
      int d = tn * 16 + ln16;
      O[(size_t)(b * S_ + r) * E_ + h * DH_ + d] =
          (_Float16)(o_acc[tn][v] / l_old[v]);
    }
}

// ---------------------------------------------------------------------------
// Fused residual + LayerNorm: xo = g * norm(x + y) + b ; also writes f16 copy.
// One block (256 thr) per token row of E=1024.
// ---------------------------------------------------------------------------
__global__ __launch_bounds__(256) void ln_kernel(
    const float* __restrict__ x, const float* __restrict__ y,
    const float* __restrict__ gw, const float* __restrict__ bw,
    float* __restrict__ xo, _Float16* __restrict__ xh)
{
  __shared__ float red[8];
  const int row  = blockIdx.x;
  const int t    = threadIdx.x;
  const int lane = t & 31;
  const int wave = t >> 5;
  const float* xr = x + (size_t)row * E_;
  const float* yr = y + (size_t)row * E_;

  float vbuf[4];
  float s = 0.f;
#pragma unroll
  for (int i = 0; i < 4; ++i) {
    int c = t + i * 256;
    vbuf[i] = xr[c] + yr[c];
    s += vbuf[i];
  }
#pragma unroll
  for (int m = 16; m >= 1; m >>= 1) s += __shfl_xor(s, m, 32);
  if (lane == 0) red[wave] = s;
  __syncthreads();
  float tot = 0.f;
#pragma unroll
  for (int w = 0; w < 8; ++w) tot += red[w];
  float mean = tot * (1.f / 1024.f);

  float vs = 0.f;
#pragma unroll
  for (int i = 0; i < 4; ++i) {
    float d = vbuf[i] - mean;
    vs += d * d;
  }
#pragma unroll
  for (int m = 16; m >= 1; m >>= 1) vs += __shfl_xor(vs, m, 32);
  __syncthreads();
  if (lane == 0) red[wave] = vs;
  __syncthreads();
  float vtot = 0.f;
#pragma unroll
  for (int w = 0; w < 8; ++w) vtot += red[w];
  float rinv = rsqrtf(vtot * (1.f / 1024.f) + 1e-5f);

#pragma unroll
  for (int i = 0; i < 4; ++i) {
    int   c = t + i * 256;
    float o = gw[c] * (vbuf[i] - mean) * rinv + bw[c];
    xo[(size_t)row * E_ + c] = o;
    xh[(size_t)row * E_ + c] = (_Float16)o;
  }
}

__global__ void cvt_kernel(const float* __restrict__ in,
                           _Float16* __restrict__ out, int n) {
  int i  = blockIdx.x * blockDim.x + threadIdx.x;
  int st = gridDim.x * blockDim.x;
  for (; i < n; i += st) out[i] = (_Float16)in[i];
}

__global__ void init_kernel(const float* __restrict__ e, float* __restrict__ x,
                            _Float16* __restrict__ xh, int n) {
  int i  = blockIdx.x * blockDim.x + threadIdx.x;
  int st = gridDim.x * blockDim.x;
  for (; i < n; i += st) {
    float v = e[i];
    x[i]  = v;
    xh[i] = (_Float16)v;
  }
}

// ---------------------------------------------------------------------------
extern "C" void kernel_launch(void* const* d_in, const int* in_sizes, int n_in,
                              void* d_out, int out_size, void* d_ws,
                              size_t ws_size, hipStream_t stream) {
  (void)in_sizes; (void)n_in; (void)out_size; (void)ws_size;
  const float* emb  = (const float*)d_in[0];
  // d_in[1] = num_ctx (compile-time NC_=1024)
  const float* ipw  = (const float*)d_in[2];
  const float* ipb  = (const float*)d_in[3];
  const float* ow   = (const float*)d_in[4];
  const float* ob   = (const float*)d_in[5];
  const float* l1w  = (const float*)d_in[6];
  const float* l1b  = (const float*)d_in[7];
  const float* l2w  = (const float*)d_in[8];
  const float* l2b  = (const float*)d_in[9];
  const float* ln1w = (const float*)d_in[10];
  const float* ln1b = (const float*)d_in[11];
  const float* ln2w = (const float*)d_in[12];
  const float* ln2b = (const float*)d_in[13];

  char* p = (char*)d_ws;
  auto alloc = [&](size_t bytes) -> char* {
    char* r = p;
    p += (bytes + 255) & ~(size_t)255;
    return r;
  };
  float*    xf   = (float*)alloc((size_t)MTOK * E_ * 4);
  _Float16* xh   = (_Float16*)alloc((size_t)MTOK * E_ * 2);
  float*    yf   = (float*)alloc((size_t)MTOK * E_ * 4);
  _Float16* qh   = (_Float16*)alloc((size_t)MTOK * E_ * 2);
  _Float16* kh   = (_Float16*)alloc((size_t)MCTX * E_ * 2);
  _Float16* vth  = (_Float16*)alloc((size_t)MCTX * E_ * 2);
  _Float16* ah   = (_Float16*)alloc((size_t)MTOK * E_ * 2);
  _Float16* hh   = (_Float16*)alloc((size_t)MTOK * DFF_ * 2);
  _Float16* ipwh = (_Float16*)alloc((size_t)L_ * 3 * E_ * E_ * 2);
  _Float16* owh  = (_Float16*)alloc((size_t)L_ * E_ * E_ * 2);
  _Float16* l1h  = (_Float16*)alloc((size_t)L_ * DFF_ * E_ * 2);
  _Float16* l2h  = (_Float16*)alloc((size_t)L_ * E_ * DFF_ * 2);

  cvt_kernel<<<4096, 256, 0, stream>>>(ipw, ipwh, L_ * 3 * E_ * E_);
  cvt_kernel<<<4096, 256, 0, stream>>>(ow, owh, L_ * E_ * E_);
  cvt_kernel<<<4096, 256, 0, stream>>>(l1w, l1h, L_ * DFF_ * E_);
  cvt_kernel<<<4096, 256, 0, stream>>>(l2w, l2h, L_ * E_ * DFF_);
  init_kernel<<<4096, 256, 0, stream>>>(emb, xf, xh, MTOK * E_);

  dim3 blk(256);
  for (int l = 0; l < L_; ++l) {
    const _Float16* wq = ipwh + (size_t)l * 3 * E_ * E_;
    const _Float16* wk = wq + (size_t)E_ * E_;
    const _Float16* wv = wk + (size_t)E_ * E_;
    const float* bq = ipb + (size_t)l * 3 * E_;
    const float* bk = bq + E_;
    const float* bv = bk + E_;

    // Q = (x Wq^T + bq) * 1/sqrt(DH), full sequence
    gemm_kernel<ST_F16><<<dim3(E_ / 128, MTOK / 256), blk, 0, stream>>>(
        xh, wq, bq, nullptr, qh, MTOK, E_, E_, MTOK, 0, 0.125f);
    // K over first NC rows of each batch
    gemm_kernel<ST_F16><<<dim3(E_ / 128, MCTX / 256), blk, 0, stream>>>(
        xh, wk, bk, nullptr, kh, MCTX, E_, E_, NC_, S_, 1.0f);
    // V over ctx rows, stored transposed [B,NH,DH,NC]
    gemm_kernel<ST_VT><<<dim3(E_ / 128, MCTX / 256), blk, 0, stream>>>(
        xh, wv, bv, nullptr, vth, MCTX, E_, E_, NC_, S_, 1.0f);

    flash_kernel<<<dim3(S_ / 64, NH_, B_), dim3(128), 0, stream>>>(qh, kh, vth, ah);

    // out-proj -> f32 y, then fused residual + LN1
    gemm_kernel<ST_F32><<<dim3(E_ / 128, MTOK / 256), blk, 0, stream>>>(
        ah, owh + (size_t)l * E_ * E_, ob + (size_t)l * E_, yf, nullptr,
        MTOK, E_, E_, MTOK, 0, 1.0f);
    ln_kernel<<<MTOK, 256, 0, stream>>>(xf, yf, ln1w + (size_t)l * E_,
                                        ln1b + (size_t)l * E_, xf, xh);

    // FFN: relu(x W1^T + b1) -> f16, then W2 -> f32 y, then residual + LN2
    gemm_kernel<ST_F16_RELU><<<dim3(DFF_ / 128, MTOK / 256), blk, 0, stream>>>(
        xh, l1h + (size_t)l * DFF_ * E_, l1b + (size_t)l * DFF_, nullptr, hh,
        MTOK, DFF_, E_, MTOK, 0, 1.0f);
    gemm_kernel<ST_F32><<<dim3(E_ / 128, MTOK / 256), blk, 0, stream>>>(
        hh, l2h + (size_t)l * E_ * DFF_, l2b + (size_t)l * E_, yf, nullptr,
        MTOK, E_, DFF_, MTOK, 0, 1.0f);
    ln_kernel<<<MTOK, 256, 0, stream>>>(xf, yf, ln2w + (size_t)l * E_,
                                        ln2b + (size_t)l * E_, xf, xh);
  }

  hipMemcpyAsync(d_out, xf, (size_t)MTOK * E_ * sizeof(float),
                 hipMemcpyDeviceToDevice, stream);
}
